// LSTMLayer_30915174596727
// MI455X (gfx1250) — matrix operand order repaired and checked by reading, not compiled, not run on hardware
//
#include <hip/hip_runtime.h>
#include <math.h>

// LSTM: B=64, T=256, D=600, H=1024. Gate order i,f,g,o (PyTorch).
#define BSZ  64
#define TLEN 256
#define DIN  600
#define HDIM 1024
#define G4H  4096
#define KP   608   // D padded to a multiple of 32 for the K loop

typedef __attribute__((ext_vector_type(16))) __bf16        v16bf;
typedef __attribute__((ext_vector_type(8)))  float         v8f;
typedef __attribute__((ext_vector_type(4)))  unsigned int  u32x4;
typedef __attribute__((ext_vector_type(8)))  unsigned int  u32x8;

// ---------------------------------------------------------------- helpers

__device__ __forceinline__ unsigned short f2bf(float f) {
  unsigned int u = __builtin_bit_cast(unsigned int, f);
  u += 0x7FFFu + ((u >> 16) & 1u);     // round-to-nearest-even
  return (unsigned short)(u >> 16);
}

__device__ __forceinline__ float sigmoidf_(float x) {
  return 1.0f / (1.0f + __expf(-x));
}

// A fragment, 16x32 bf16 (MxK), row-major source with leading dim lda.
// Layout: lanes 0-15 hold M=0..15; elems 0..15 -> K = (lanehi?16:0)+0..15
// (contiguous, matching load_b_frag).
__device__ __forceinline__ v16bf load_a_frag(const unsigned short* __restrict__ A,
                                             int lda, int m0, int k0, int lane) {
  int m  = m0 + (lane & 15);
  int hi = (lane & 16) ? 16 : 0;
  const unsigned short* p = A + (size_t)m * lda + k0 + hi;
  u32x4 lo  = *(const u32x4*)(p);
  u32x4 hi4 = *(const u32x4*)(p + 8);
  u32x8 u;
  u[0] = lo[0];  u[1] = lo[1];  u[2] = lo[2];  u[3] = lo[3];
  u[4] = hi4[0]; u[5] = hi4[1]; u[6] = hi4[2]; u[7] = hi4[3];
  return __builtin_bit_cast(v16bf, u);
}

// B fragment, 32x16 bf16 (KxN), with B[k][n] = W[n][k] (W row-major, ld ldb).
// Contiguous-K layout used here (as in load_a_frag): lane n = lane&15; lanes 0-15 hold K=k0..k0+15, lanes 16-31 hold
// K=k0+16..k0+31 -> one contiguous 32-byte load per lane from W's row n.
__device__ __forceinline__ v16bf load_b_frag(const unsigned short* __restrict__ W,
                                             int ldb, int n0, int k0, int lane) {
  int n    = n0 + (lane & 15);
  int koff = (lane & 16) ? 16 : 0;
  const unsigned short* p = W + (size_t)n * ldb + k0 + koff;
  u32x8 u = *(const u32x8*)p;
  return __builtin_bit_cast(v16bf, u);
}

// ---------------------------------------------------------------- conversion

// Convert fp32 [rows][scols] -> bf16 [rows][dcols], zero-padding cols >= scols.
__global__ void cvt_pad_bf16_kernel(const float* __restrict__ src,
                                    unsigned short* __restrict__ dst,
                                    int rows, int scols, int dcols) {
  size_t idx   = (size_t)blockIdx.x * blockDim.x + threadIdx.x;
  size_t total = (size_t)rows * dcols;
  if (idx >= total) return;
  int    col = (int)(idx % dcols);
  size_t row = idx / dcols;
  float v = (col < scols) ? src[row * (size_t)scols + col] : 0.0f;
  dst[idx] = f2bf(v);
}

__global__ void init_state_kernel(const float* __restrict__ h0,
                                  const float* __restrict__ c0,
                                  unsigned short* __restrict__ hbuf,
                                  float* __restrict__ cstate) {
  int idx = blockIdx.x * blockDim.x + threadIdx.x;
  if (idx >= BSZ * HDIM) return;
  hbuf[idx]   = f2bf(h0[idx]);
  cstate[idx] = c0[idx];
}

// ---------------------------------------------------------------- phase 1
// x_gates[t][b][4H] = obs[b][t][:] @ w_ih^T + b_ih + b_hh
// GEMM: M = B*T = 16384 (row = b*T + t), N = 4096, K = 608 (padded).
// Block = 256 threads (8 waves), 64x64 output tile; wave = 16(M) x 32(N).
__global__ __launch_bounds__(256)
void xproj_kernel(const unsigned short* __restrict__ Abf,   // [16384][KP] bf16
                  const unsigned short* __restrict__ Wbf,   // [4096][KP]  bf16
                  const float* __restrict__ b_ih,
                  const float* __restrict__ b_hh,
                  float* __restrict__ xg)                   // [T][B][4H] f32
{
  int lane = threadIdx.x & 31;
  int wave = threadIdx.x >> 5;
  int m0 = blockIdx.y * 64 + (wave >> 1) * 16;
  int nb = blockIdx.x * 64 + (wave & 1) * 32;

  v8f acc0 = {}; v8f acc1 = {};
  for (int k0 = 0; k0 < KP; k0 += 32) {
    v16bf a  = load_a_frag(Abf, KP, m0, k0, lane);
    v16bf b0 = load_b_frag(Wbf, KP, nb,      k0, lane);
    v16bf b1 = load_b_frag(Wbf, KP, nb + 16, k0, lane);
    acc0 = __builtin_amdgcn_wmma_f32_16x16x32_bf16(false, a, false, b0,
                                                   (short)0, acc0, false, false);
    acc1 = __builtin_amdgcn_wmma_f32_16x16x32_bf16(false, a, false, b1,
                                                   (short)0, acc1, false, false);
  }

  int rowbase = m0 + ((lane & 16) ? 8 : 0);
  int ncol    = nb + (lane & 15);
#pragma unroll
  for (int r = 0; r < 8; ++r) {
    int gr = rowbase + r;            // global row = b*T + t
    int b_ = gr >> 8;                // / TLEN
    int t_ = gr & (TLEN - 1);
    size_t base = (size_t)(t_ * BSZ + b_) * G4H;
    xg[base + ncol]      = acc0[r] + b_ih[ncol]      + b_hh[ncol];
    xg[base + ncol + 16] = acc1[r] + b_ih[ncol + 16] + b_hh[ncol + 16];
  }
}

// ---------------------------------------------------------------- phase 2
// One timestep: gates = xg[t] + h_prev @ w_hh^T; fused cell update.
// Grid = H/32 = 32 blocks x 512 threads (16 waves). Each block owns 32 h-cols.
// K=1024 is SPLIT IN TWO across the wave halves to shorten the dependent WMMA
// chain on the sequential critical path:
//   waves 0..7  : K = [0,512),   seeded with xg (input projection)
//   waves 8..15 : K = [512,1024), seeded with zero; partials -> LDS
// After one barrier, waves 0..7 reduce and run the fused cell update.
// Each wave computes the SAME 16x16 (batch x col) tile for all 4 gates, so
// i,f,g,o for an element live in the same lane/slot -> no shuffles needed.
__global__ __launch_bounds__(512)
void lstm_step_kernel(const float* __restrict__ xg,           // [T][B][4H]
                      const unsigned short* __restrict__ Whh, // [4H][H] bf16
                      const unsigned short* __restrict__ h_in,// [B][H]  bf16
                      unsigned short* __restrict__ h_out,     // [B][H]  bf16
                      float* __restrict__ cstate,             // [B][H]  f32
                      float* __restrict__ out,                // [B][T][H]
                      float* __restrict__ h_n,                // [B][H]
                      float* __restrict__ c_n,                // [B][H]
                      int t)
{
  __shared__ float red[8 * 4 * 32 * 8];         // 32 KB partial accumulators

  int lane   = threadIdx.x & 31;
  int wave   = threadIdx.x >> 5;
  int kgrp   = wave >> 3;                       // 0: low half-K, 1: high half-K
  int wsub   = wave & 7;
  int m0 = (wsub >> 1) * 16;                    // batch rows (B=64 -> 4 tiles)
  int n0 = blockIdx.x * 32 + (wsub & 1) * 16;   // h column base

  int rowbase = m0 + ((lane & 16) ? 8 : 0);
  int ncol    = n0 + (lane & 15);

  // Seed: low-K waves carry the input projection, high-K waves start at zero.
  v8f acc[4];
  if (kgrp == 0) {
    const float* xgt = xg + (size_t)t * BSZ * G4H;
#pragma unroll
    for (int g = 0; g < 4; ++g) {
      v8f a;
#pragma unroll
      for (int r = 0; r < 8; ++r)
        a[r] = xgt[(size_t)(rowbase + r) * G4H + g * HDIM + ncol];
      acc[g] = a;
    }
  } else {
#pragma unroll
    for (int g = 0; g < 4; ++g) acc[g] = (v8f){};
  }

  // gates += h_prev @ w_hh^T over this wave's half of K.
  // One A fragment is reused across the 4 gates.
  int kbeg = kgrp * (HDIM / 2);
  int kend = kbeg + (HDIM / 2);
  for (int k0 = kbeg; k0 < kend; k0 += 32) {
    v16bf a  = load_a_frag(h_in, HDIM, m0, k0, lane);
    v16bf bi = load_b_frag(Whh, HDIM, 0 * HDIM + n0, k0, lane);
    v16bf bf = load_b_frag(Whh, HDIM, 1 * HDIM + n0, k0, lane);
    v16bf bg = load_b_frag(Whh, HDIM, 2 * HDIM + n0, k0, lane);
    v16bf bo = load_b_frag(Whh, HDIM, 3 * HDIM + n0, k0, lane);
    acc[0] = __builtin_amdgcn_wmma_f32_16x16x32_bf16(false, a, false, bi,
                                                     (short)0, acc[0], false, false);
    acc[1] = __builtin_amdgcn_wmma_f32_16x16x32_bf16(false, a, false, bf,
                                                     (short)0, acc[1], false, false);
    acc[2] = __builtin_amdgcn_wmma_f32_16x16x32_bf16(false, a, false, bg,
                                                     (short)0, acc[2], false, false);
    acc[3] = __builtin_amdgcn_wmma_f32_16x16x32_bf16(false, a, false, bo,
                                                     (short)0, acc[3], false, false);
  }

  // High-K waves publish partials to LDS (contiguous 32B per lane -> b128s).
  if (kgrp == 1) {
#pragma unroll
    for (int g = 0; g < 4; ++g) {
      float* dst = &red[(((wsub * 4 + g) * 32) + lane) * 8];
#pragma unroll
      for (int r = 0; r < 8; ++r) dst[r] = acc[g][r];
    }
  }
  __syncthreads();
  if (kgrp == 1) return;

  // Low-K waves: reduce the two half-K partials, then fused cell update.
#pragma unroll
  for (int g = 0; g < 4; ++g) {
    const float* src = &red[(((wsub * 4 + g) * 32) + lane) * 8];
#pragma unroll
    for (int r = 0; r < 8; ++r) acc[g][r] += src[r];
  }

#pragma unroll
  for (int r = 0; r < 8; ++r) {
    int b_ = rowbase + r;                       // batch index
    float iv = sigmoidf_(acc[0][r]);
    float fv = sigmoidf_(acc[1][r]);
    float gv = tanhf(acc[2][r]);
    float ov = sigmoidf_(acc[3][r]);
    size_t sidx = (size_t)b_ * HDIM + ncol;
    float cp = cstate[sidx];
    float cn = fv * cp + iv * gv;
    float hn = ov * tanhf(cn);
    cstate[sidx] = cn;
    h_out[sidx]  = f2bf(hn);
    out[((size_t)b_ * TLEN + t) * HDIM + ncol] = hn;
    if (t == TLEN - 1) { h_n[sidx] = hn; c_n[sidx] = cn; }
  }
}

// ---------------------------------------------------------------- launch

extern "C" void kernel_launch(void* const* d_in, const int* in_sizes, int n_in,
                              void* d_out, int out_size, void* d_ws, size_t ws_size,
                              hipStream_t stream) {
  const float* obs  = (const float*)d_in[0];   // [B,T,D]
  const float* h0   = (const float*)d_in[1];   // [1,B,H]
  const float* c0   = (const float*)d_in[2];   // [1,B,H]
  const float* w_ih = (const float*)d_in[3];   // [4H,D]
  const float* w_hh = (const float*)d_in[4];   // [4H,H]
  const float* b_ih = (const float*)d_in[5];   // [4H]
  const float* b_hh = (const float*)d_in[6];   // [4H]

  float* out = (float*)d_out;                            // [B,T,H]
  float* h_n = out + (size_t)BSZ * TLEN * HDIM;          // [B,H]
  float* c_n = h_n + (size_t)BSZ * HDIM;                 // [B,H]

  // Workspace layout (all chunks 256B-aligned by construction).
  char* ws = (char*)d_ws;
  size_t off = 0;
  float* xg = (float*)(ws + off);                 off += (size_t)TLEN * BSZ * G4H * 4;
  unsigned short* obs_bf = (unsigned short*)(ws + off); off += (size_t)BSZ * TLEN * KP * 2;
  unsigned short* wih_bf = (unsigned short*)(ws + off); off += (size_t)G4H * KP * 2;
  unsigned short* whh_bf = (unsigned short*)(ws + off); off += (size_t)G4H * HDIM * 2;
  unsigned short* hbuf0  = (unsigned short*)(ws + off); off += (size_t)BSZ * HDIM * 2;
  unsigned short* hbuf1  = (unsigned short*)(ws + off); off += (size_t)BSZ * HDIM * 2;
  float* cstate = (float*)(ws + off);             off += (size_t)BSZ * HDIM * 4;
  (void)ws_size; (void)in_sizes; (void)n_in; (void)out_size;

  // bf16 conversions (obs & w_ih K-padded 600 -> 608).
  {
    size_t n = (size_t)BSZ * TLEN * KP;
    cvt_pad_bf16_kernel<<<(unsigned)((n + 255) / 256), 256, 0, stream>>>(
        obs, obs_bf, BSZ * TLEN, DIN, KP);
  }
  {
    size_t n = (size_t)G4H * KP;
    cvt_pad_bf16_kernel<<<(unsigned)((n + 255) / 256), 256, 0, stream>>>(
        w_ih, wih_bf, G4H, DIN, KP);
  }
  {
    size_t n = (size_t)G4H * HDIM;
    cvt_pad_bf16_kernel<<<(unsigned)((n + 255) / 256), 256, 0, stream>>>(
        w_hh, whh_bf, G4H, HDIM, HDIM);
  }
  init_state_kernel<<<(BSZ * HDIM + 255) / 256, 256, 0, stream>>>(
      h0, c0, hbuf0, cstate);

  // Phase 1: big parallel GEMM for all timesteps' input projections.
  {
    dim3 grid(G4H / 64, (BSZ * TLEN) / 64);
    xproj_kernel<<<grid, 256, 0, stream>>>(obs_bf, wih_bf, b_ih, b_hh, xg);
  }

  // Phase 2: sequential scan; stream order enforces the recurrence,
  // h double-buffered so readers of h_prev never race writers of h_new.
  for (int t = 0; t < TLEN; ++t) {
    const unsigned short* hi = (t & 1) ? hbuf1 : hbuf0;
    unsigned short*       ho = (t & 1) ? hbuf0 : hbuf1;
    lstm_step_kernel<<<HDIM / 32, 512, 0, stream>>>(
        xg, whh_bf, hi, ho, cstate, out, h_n, c_n, t);
  }
}
